// CharRNN_11862699672000
// MI455X (gfx1250) — compile-verified, matrix operand order checked by reading, not run on hardware
//
#include <hip/hip_runtime.h>
#include <hip/hip_bf16.h>
#include <cstdint>

// ---------------------------------------------------------------------------
// CharRNN forward for gfx1250 (MI455X): f16 WMMA GEMMs + persistent scan.
//   xp   = x @ W_ih^T + (b_ih + b_hh)            [B,T,H]  (one GEMM)
//   h_t  = tanh(xp_t + h_{t-1} @ W_hh^T)         persistent kernel, T steps
//   out  = hseq @ W_fc^T + b_fc                  [B,T,V]  (one GEMM)
// Matrix math: v_wmma_f32_16x16x32_f16 (fp32 accumulate).
// W slices staged global->LDS once via GLOBAL_LOAD_ASYNC_TO_LDS_B128
// (ASYNCcnt); scan steps sync with a device-scope atomic barrier.
// ---------------------------------------------------------------------------

typedef _Float16 f16;
typedef __attribute__((ext_vector_type(16))) _Float16     v16h;
typedef __attribute__((ext_vector_type(8)))  float        v8f;
typedef __attribute__((ext_vector_type(4)))  unsigned int v4u;

constexpr int  kB = 128;
constexpr int  kT = 1024;
constexpr int  kV = 96;
constexpr int  kH = 512;
constexpr long kM = (long)kB * kT;   // 131072 rows of x / logits
constexpr int  kScanBlocks = 8;      // persistent scan grid

// Load one 16x32 f16 fragment (A-layout; B fragments use the same pattern
// because B-columns == W-rows for C = X * W^T).
//  lanes 0-15 : row = row0+lane,    K = k0+[0..7]  (v0..3), k0+[16..23] (v4..7)
//  lanes16-31 : row = row0+lane-16, K = k0+[8..15] (v0..3), k0+[24..31] (v4..7)
// Works for global or LDS base pointers (addrspace inferred after inlining:
// global_load_b128 / ds_load_b128).
__device__ __forceinline__ v16h load_frag(const f16* base, int row0, int ld, int k0) {
  const int lane = threadIdx.x & 31;
  const int half = lane >> 4;
  const f16* p = base + (size_t)(row0 + (lane & 15)) * ld + k0 + half * 8;
  union { v16h v; v4u q[2]; } u;
  u.q[0] = *(const v4u*)p;         // 8 f16, 16B aligned
  u.q[1] = *(const v4u*)(p + 16);  // next 8 f16 (+32 bytes)
  return u.v;
}

__device__ __forceinline__ v8f wmma16(v16h a, v16h b, v8f c) {
  return __builtin_amdgcn_wmma_f32_16x16x32_f16(
      /*neg_a=*/false, a, /*neg_b=*/false, b,
      /*c_mod=*/(short)0, c, /*reuse_a=*/false, /*reuse_b=*/false);
}

// Async copy 16 bytes global -> LDS (GVS mode: SGPR64 base + VGPR32 offset).
// Tracked by ASYNCcnt; pair with s_wait_asynccnt + __syncthreads().
__device__ __forceinline__ void async_b128_to_lds(unsigned lds_byte_addr,
                                                  const f16* gbase,
                                                  unsigned gbyte_off) {
  asm volatile("global_load_async_to_lds_b128 %0, %1, %2"
               :: "v"(lds_byte_addr), "v"(gbyte_off), "s"(gbase)
               : "memory");
}

__device__ __forceinline__ void wait_async_all() {
  asm volatile("s_wait_asynccnt 0x0" ::: "memory");
}

// Device-scope split barrier for a small co-resident grid. Counter increases
// monotonically (kScanBlocks per step) -> no reset/sense needed.
__device__ __forceinline__ void grid_barrier(unsigned* bar, unsigned expected) {
  __threadfence();                       // release this thread's h stores
  __syncthreads();                       // whole block stored + fenced
  if (threadIdx.x == 0) {
    __hip_atomic_fetch_add(bar, 1u, __ATOMIC_RELEASE, __HIP_MEMORY_SCOPE_AGENT);
    while (__hip_atomic_load(bar, __ATOMIC_ACQUIRE, __HIP_MEMORY_SCOPE_AGENT)
           < expected) {
      __builtin_amdgcn_s_sleep(1);
    }
  }
  __syncthreads();
  __builtin_amdgcn_fence(__ATOMIC_ACQUIRE, "agent");   // see other blocks' h
}

// ---------------------------------------------------------------------------
// prep: f32 -> f16 conversions + fused bias + barrier reset
// ---------------------------------------------------------------------------
__global__ void __launch_bounds__(256) prep_kernel(
    const float* __restrict__ x,   const float* __restrict__ hid,
    const float* __restrict__ Wih, const float* __restrict__ Whh,
    const float* __restrict__ Wfc, const float* __restrict__ bih,
    const float* __restrict__ bhh,
    f16* __restrict__ xb,  f16* __restrict__ h0b, f16* __restrict__ wihb,
    f16* __restrict__ whhb, f16* __restrict__ wfcb, float* __restrict__ bias,
    unsigned* __restrict__ bar) {
  const long stride = (long)gridDim.x * blockDim.x;
  const long i0 = (long)blockIdx.x * blockDim.x + threadIdx.x;
  for (long i = i0; i < kM * kV;        i += stride) xb[i]   = (f16)x[i];
  for (long i = i0; i < (long)kH * kV;  i += stride) wihb[i] = (f16)Wih[i];
  for (long i = i0; i < (long)kH * kH;  i += stride) whhb[i] = (f16)Whh[i];
  for (long i = i0; i < (long)kV * kH;  i += stride) wfcb[i] = (f16)Wfc[i];
  for (long i = i0; i < (long)kB * kH;  i += stride) h0b[i]  = (f16)hid[i];
  for (long i = i0; i < kH;             i += stride) bias[i] = bih[i] + bhh[i];
  if (i0 == 0) *bar = 0u;   // reset scan barrier every launch (replay-safe)
}

// ---------------------------------------------------------------------------
// xproj: xp[M,H] = xb[M,V] @ wih[H,V]^T + bias   (M=131072, N=512, K=96)
// Each wave: 16 rows x 64 cols (4 N-tiles), 3 K-steps. 8192 blocks x 8 waves.
// Bound by the 268MB xp write -> direct global operand loads are fine.
// ---------------------------------------------------------------------------
__global__ void __launch_bounds__(256) xproj_kernel(
    const f16* __restrict__ xb, const f16* __restrict__ wih,
    const float* __restrict__ bias, float* __restrict__ xp) {
  const int wave  = (int)((blockIdx.x * blockDim.x + threadIdx.x) >> 5);
  const int lane  = threadIdx.x & 31;
  const int m0 = (wave >> 3) * 16;
  const int n0 = (wave & 7) * 64;
  const int col   = lane & 15;
  const int rbase = (lane >> 4) * 8;

  v8f acc[4];
#pragma unroll
  for (int j = 0; j < 4; ++j) {
    const float bv = bias[n0 + j * 16 + col];
#pragma unroll
    for (int r = 0; r < 8; ++r) acc[j][r] = bv;
  }
#pragma unroll
  for (int kk = 0; kk < 3; ++kk) {
    v16h a = load_frag(xb, m0, kV, kk * 32);
#pragma unroll
    for (int j = 0; j < 4; ++j) {
      v16h b = load_frag(wih, n0 + j * 16, kV, kk * 32);
      acc[j] = wmma16(a, b, acc[j]);
    }
  }
#pragma unroll
  for (int j = 0; j < 4; ++j)
#pragma unroll
    for (int r = 0; r < 8; ++r)
      xp[(size_t)(m0 + rbase + r) * kH + n0 + j * 16 + col] = acc[j][r];
}

// ---------------------------------------------------------------------------
// scan (persistent): 8 blocks; block owns 64 hidden cols. Its W_hh slice
// (64x512 f16 = 64KB) is async-staged into LDS ONCE and reused for all T
// steps. 8 waves = 8 batch tiles. Per step: acc <- xp slice, A fragments
// (h_prev) preloaded in half-K batches of 8 (one batched global latency),
// B fragments from LDS, tanh epilogue, then device-scope barrier.
// ---------------------------------------------------------------------------
__global__ void __launch_bounds__(256) scan_kernel(
    const f16* __restrict__ h0b, const f16* __restrict__ whh,
    const float* __restrict__ xp, f16* __restrict__ hseq,
    float* __restrict__ hlast, unsigned* __restrict__ bar) {
  __shared__ f16 sW[64 * kH];                       // 64KB W_hh slice
  const int tid = threadIdx.x;
  const int n0  = blockIdx.x * 64;                  // block's hidden-col base

  {  // one-time async stage: 64 contiguous rows -> 4096 x b128, 16/thread
    const unsigned lds_base = (unsigned)(uintptr_t)(&sW[0]);
    const f16* gsrc = whh + (size_t)n0 * kH;        // rows are contiguous
#pragma unroll
    for (int c = 0; c < 16; ++c) {
      const unsigned off = (unsigned)((c * 256 + tid) * 16);
      async_b128_to_lds(lds_base + off, gsrc, off);
    }
    wait_async_all();
    __syncthreads();
  }

  const int lane  = tid & 31;
  const int b0    = (tid >> 5) * 16;                // batch tile per wave
  const int col   = lane & 15;
  const int rbase = (lane >> 4) * 8;

  for (int t = 0; t < kT; ++t) {
    const f16* hprev = (t == 0) ? h0b : hseq + (size_t)(t - 1) * kB * kH;
    f16* hnext = hseq + (size_t)t * kB * kH;

    v8f acc[4];
#pragma unroll
    for (int j = 0; j < 4; ++j)
#pragma unroll
      for (int r = 0; r < 8; ++r)
        acc[j][r] = xp[((size_t)(b0 + rbase + r) * kT + t) * kH + n0 + j * 16 + col];

#pragma unroll
    for (int hk = 0; hk < 2; ++hk) {                // K halves: 8 frags each
      v16h afr[8];
#pragma unroll
      for (int kk = 0; kk < 8; ++kk)                // batched A loads (L2)
        afr[kk] = load_frag(hprev, b0, kH, (hk * 8 + kk) * 32);
#pragma unroll
      for (int kk = 0; kk < 8; ++kk)
#pragma unroll
        for (int j = 0; j < 4; ++j) {
          v16h b = load_frag(sW, j * 16, kH, (hk * 8 + kk) * 32);  // LDS
          acc[j] = wmma16(afr[kk], b, acc[j]);
        }
    }

#pragma unroll
    for (int j = 0; j < 4; ++j)
#pragma unroll
      for (int r = 0; r < 8; ++r) {
        const size_t idx = (size_t)(b0 + rbase + r) * kH + n0 + j * 16 + col;
        const float v = tanhf(acc[j][r]);
        hnext[idx] = (f16)v;
        if (t == kT - 1) hlast[idx] = v;            // uniform branch
      }

    grid_barrier(bar, (unsigned)(kScanBlocks * (t + 1)));
  }
}

// ---------------------------------------------------------------------------
// fc: out[b,t,v] = hseq[t,b,:] @ wfc[v,:] + b_fc   (M=T*B, N=96, K=512)
// 1024 blocks (one timestep each) x 8 waves (batch tiles). W_fc staged in
// LDS in two 48KB K-halves (96 rows x 256 K) via async copies.
// ---------------------------------------------------------------------------
__global__ void __launch_bounds__(256) fc_kernel(
    const f16* __restrict__ hseq, const f16* __restrict__ wfc,
    const float* __restrict__ bfc, float* __restrict__ out) {
  __shared__ f16 sW[96 * 256];                      // 48KB K-half of W_fc
  const int tid   = threadIdx.x;
  const int lane  = tid & 31;
  const int t     = blockIdx.x;
  const int b0    = (tid >> 5) * 16;
  const int row0  = t * kB + b0;                    // row in hseq [T*B, H]
  const int col   = lane & 15;
  const int rbase = (lane >> 4) * 8;
  const unsigned lds_base = (unsigned)(uintptr_t)(&sW[0]);

  v8f acc[6];
#pragma unroll
  for (int j = 0; j < 6; ++j) {
    const float bv = bfc[j * 16 + col];
#pragma unroll
    for (int r = 0; r < 8; ++r) acc[j][r] = bv;
  }

  for (int h = 0; h < 2; ++h) {                     // K-halves: [0,256), [256,512)
    // stage wfc[0:96, h*256 : h*256+256): 3072 x b128, 12 per thread
#pragma unroll
    for (int c = 0; c < 12; ++c) {
      const int i = c * 256 + tid;                  // b128 index
      const int wrow = i >> 5;                      // 32 chunks per 256-K row
      const int wchk = i & 31;
      const unsigned goff = (unsigned)(((size_t)wrow * kH + h * 256 + wchk * 8) * 2);
      async_b128_to_lds(lds_base + (unsigned)(i * 16), wfc, goff);
    }
    wait_async_all();
    __syncthreads();

#pragma unroll 4
    for (int kk = 0; kk < 8; ++kk) {
      v16h a = load_frag(hseq, row0, kH, h * 256 + kk * 32);
#pragma unroll
      for (int j = 0; j < 6; ++j) {
        v16h b = load_frag(sW, j * 16, 256, kk * 32);   // LDS, ld=256
        acc[j] = wmma16(a, b, acc[j]);
      }
    }
    __syncthreads();                                // before overwriting sW
  }

#pragma unroll
  for (int j = 0; j < 6; ++j)
#pragma unroll
    for (int r = 0; r < 8; ++r)
      out[((size_t)(b0 + rbase + r) * kT + t) * kV + j * 16 + col] = acc[j][r];
}

// ---------------------------------------------------------------------------
extern "C" void kernel_launch(void* const* d_in, const int* in_sizes, int n_in,
                              void* d_out, int out_size, void* d_ws, size_t ws_size,
                              hipStream_t stream) {
  const float* x   = (const float*)d_in[0];   // [B,T,V]
  const float* hid = (const float*)d_in[1];   // [1,B,H]
  const float* Wih = (const float*)d_in[2];   // [H,V]
  const float* Whh = (const float*)d_in[3];   // [H,H]
  const float* bih = (const float*)d_in[4];   // [H]
  const float* bhh = (const float*)d_in[5];   // [H]
  const float* Wfc = (const float*)d_in[6];   // [V,H]
  const float* bfc = (const float*)d_in[7];   // [V]
  float* out = (float*)d_out;                 // logits [B,T,V] ++ h_last [1,B,H]

  char* ws = (char*)d_ws;
  size_t off = 0;
  auto carve = [&](size_t bytes) -> void* {
    void* p = ws + off;
    off += (bytes + 255) & ~(size_t)255;
    return p;
  };
  f16*      xb   = (f16*)     carve((size_t)kM * kV * 2);       //  25 MB
  f16*      wihb = (f16*)     carve((size_t)kH * kV * 2);
  f16*      whhb = (f16*)     carve((size_t)kH * kH * 2);       // 512 KB
  f16*      wfcb = (f16*)     carve((size_t)kV * kH * 2);
  float*    bias = (float*)   carve((size_t)kH * 4);
  f16*      h0b  = (f16*)     carve((size_t)kB * kH * 2);
  unsigned* bar  = (unsigned*)carve(256);
  float*    xp   = (float*)   carve((size_t)kM * kH * 4);       // 268 MB
  f16*      hseq = (f16*)     carve((size_t)kT * kB * kH * 2);  // 134 MB
  (void)ws_size; (void)in_sizes; (void)n_in; (void)out_size;

  prep_kernel<<<4096, 256, 0, stream>>>(x, hid, Wih, Whh, Wfc, bih, bhh,
                                        xb, h0b, wihb, whhb, wfcb, bias, bar);

  // xp = x @ W_ih^T + (b_ih + b_hh)
  xproj_kernel<<<8192, 256, 0, stream>>>(xb, wihb, bias, xp);

  // Persistent scan over all T steps (device-scope barrier between steps).
  float* hlast = out + (size_t)kB * kT * kV;
  scan_kernel<<<kScanBlocks, 256, 0, stream>>>(h0b, whhb, xp, hseq, hlast, bar);

  // logits = hseq @ W_fc^T + b_fc  (with [T,B]->[B,T] remap in epilogue)
  fc_kernel<<<1024, 256, 0, stream>>>(hseq, wfcb, bfc, out);
}
